// SingleAttentionWithPairBias_53867479826845
// MI455X (gfx1250) — matrix operand order repaired and checked by reading, not compiled, not run on hardware
//
#include <hip/hip_runtime.h>
#include <hip/hip_bf16.h>

typedef __bf16 bf16_t;
typedef __attribute__((ext_vector_type(16))) __bf16 v16bf;
typedef __attribute__((ext_vector_type(8)))  float  v8f;

#define N_TOK 768
#define DIM   768
#define NH    16
#define DH    48
#define DP    128
#define DHP   64      // padded head dim for WMMA K=32 stepping
#define LN_EPS 1e-5f

#define MODE_Q   0
#define MODE_KT  1
#define MODE_V   2
#define MODE_G   3
#define MODE_OUT 4

union Frag16 { v16bf v; uint4 u[2]; };

__device__ __forceinline__ float rsum32(float x) {
#pragma unroll
  for (int m = 16; m >= 1; m >>= 1) x += __shfl_xor(x, m, 32);
  return x;
}
__device__ __forceinline__ float rmax16(float x) {
#pragma unroll
  for (int m = 8; m >= 1; m >>= 1) x = fmaxf(x, __shfl_xor(x, m, 32));
  return x;
}
__device__ __forceinline__ float rsum16(float x) {
#pragma unroll
  for (int m = 8; m >= 1; m >>= 1) x += __shfl_xor(x, m, 32);
  return x;
}

// out[c*R + r] = (bf16) in[r*C + c]   (transpose + convert)
__global__ void k_transpose_bf16(const float* __restrict__ in, bf16_t* __restrict__ out,
                                 int R, int C) {
  int id = blockIdx.x * blockDim.x + threadIdx.x;
  if (id >= R * C) return;
  int r = id / C, c = id % C;
  out[(size_t)c * R + r] = (bf16_t)in[(size_t)r * C + c];
}

// LayerNorm rows of s (768 wide), emit bf16. One 256-thread block per row.
__global__ void k_ln_s(const float* __restrict__ s, const float* __restrict__ w,
                       const float* __restrict__ b, bf16_t* __restrict__ sn) {
  __shared__ float sm[256], sv[256];
  int row = blockIdx.x;
  const float* x = s + (size_t)row * DIM;
  float vals[3], acc = 0.f, acc2 = 0.f;
#pragma unroll
  for (int i = 0; i < 3; i++) {
    float v = x[threadIdx.x + i * 256];
    vals[i] = v; acc += v; acc2 += v * v;
  }
  sm[threadIdx.x] = acc; sv[threadIdx.x] = acc2;
  __syncthreads();
  for (int st = 128; st > 0; st >>= 1) {
    if (threadIdx.x < st) { sm[threadIdx.x] += sm[threadIdx.x + st]; sv[threadIdx.x] += sv[threadIdx.x + st]; }
    __syncthreads();
  }
  float mean = sm[0] / DIM;
  float var  = sv[0] / DIM - mean * mean;
  float rs   = rsqrtf(var + LN_EPS);
#pragma unroll
  for (int i = 0; i < 3; i++) {
    int j = threadIdx.x + i * 256;
    sn[(size_t)row * DIM + j] = (bf16_t)((vals[i] - mean) * rs * w[j] + b[j]);
  }
}

// Generic bf16 WMMA GEMM: C(16x16 tile per wave) = A(N x 768, bf16 row-major)
// x B(wt: 768 x 768 bf16, already transposed so B[k][j] contiguous in j).
__global__ void k_gemm(const bf16_t* __restrict__ A, const bf16_t* __restrict__ Bw,
                       const float* __restrict__ bq,
                       bf16_t* __restrict__ outb, float* __restrict__ outf, int mode) {
  int wave = (blockIdx.x * blockDim.x + threadIdx.x) >> 5;
  int lane = threadIdx.x & 31;
  const int tilesC = DIM / 16;
  int rt = wave / tilesC, ct = wave % tilesC;
  if (rt >= N_TOK / 16) return;
  int row0 = rt * 16, col0 = ct * 16;
  int m = lane & 15, half = lane >> 4;
  v8f acc = {};
#pragma unroll 4
  for (int kk = 0; kk < DIM / 32; kk++) {
    Frag16 a, b;
    // A fragment: lane=row M, K chunks {half*8..+7} and {16+half*8..+7}
    const bf16_t* ap = A + (size_t)(row0 + m) * DIM + kk * 32 + half * 8;
    a.u[0] = *(const uint4*)(ap);
    a.u[1] = *(const uint4*)(ap + 16);
    // B fragment: lane=col N = m, K chunks {half*8..+7} and {16+half*8..+7}
    const bf16_t* bp = Bw + (size_t)(kk * 32 + half * 8) * DIM + col0 + m;
    for (int s = 0; s < 8; s++) { b.v[s] = bp[(size_t)s * DIM]; b.v[8 + s] = bp[(size_t)(16 + s) * DIM]; }
    acc = __builtin_amdgcn_wmma_f32_16x16x32_bf16(false, a.v, false, b.v,
                                                  (short)0, acc, false, false);
  }
  int j = col0 + m;
  const float qscale = 0.14433756729740643f;  // 1/sqrt(48)
#pragma unroll
  for (int r = 0; r < 8; r++) {
    int n = row0 + r + 8 * half;   // C layout: row M = r + 8*half, col = m
    float v = acc[r];
    if (mode == MODE_Q) {
      int h = j / DH, dh = j % DH;
      outb[(size_t)n * (NH * DHP) + h * DHP + dh] = (bf16_t)((v + bq[j]) * qscale);
    } else if (mode == MODE_KT) {
      int h = j / DH, dh = j % DH;
      outb[((size_t)h * DHP + dh) * N_TOK + n] = (bf16_t)v;
    } else if (mode == MODE_V) {
      outb[(size_t)n * DIM + j] = (bf16_t)v;
    } else if (mode == MODE_G) {
      outf[(size_t)n * DIM + j] = 1.0f / (1.0f + __expf(-v));
    } else {
      outf[(size_t)n * DIM + j] = v;
    }
  }
}

// Pair bias: one wave handles 16 (q,k) pairs (fixed q, 16 consecutive k).
// LN over DP=128, then (16 x 128) @ (128 x 16 heads) via 4 bf16 WMMAs.
__global__ void k_pairbias(const float* __restrict__ z, const float* __restrict__ lw,
                           const float* __restrict__ lb, const bf16_t* __restrict__ wzT,
                           float* __restrict__ bias) {
  __shared__ __align__(16) bf16_t alds[8][16 * DP];
  int wv = threadIdx.x >> 5, lane = threadIdx.x & 31;
  int wave = blockIdx.x * 8 + wv;
  int q  = wave / (N_TOK / 16);
  int k0 = (wave % (N_TOK / 16)) * 16;

  float lwv[4], lbv[4];
#pragma unroll
  for (int i = 0; i < 4; i++) { lwv[i] = lw[lane * 4 + i]; lbv[i] = lb[lane * 4 + i]; }

  for (int mrow = 0; mrow < 16; mrow++) {
    const float4 zv = *(const float4*)(z + ((size_t)q * N_TOK + k0 + mrow) * DP + lane * 4);
    float s1 = zv.x + zv.y + zv.z + zv.w;
    float s2 = zv.x * zv.x + zv.y * zv.y + zv.z * zv.z + zv.w * zv.w;
    s1 = rsum32(s1); s2 = rsum32(s2);
    float mean = s1 / DP, var = s2 / DP - mean * mean;
    float rs = rsqrtf(var + LN_EPS);
    bf16_t* dst = &alds[wv][mrow * DP + lane * 4];
    dst[0] = (bf16_t)((zv.x - mean) * rs * lwv[0] + lbv[0]);
    dst[1] = (bf16_t)((zv.y - mean) * rs * lwv[1] + lbv[1]);
    dst[2] = (bf16_t)((zv.z - mean) * rs * lwv[2] + lbv[2]);
    dst[3] = (bf16_t)((zv.w - mean) * rs * lwv[3] + lbv[3]);
  }
  __syncthreads();

  int m = lane & 15, half = lane >> 4;
  v8f acc = {};
#pragma unroll
  for (int t = 0; t < 4; t++) {
    Frag16 a, b;
    const bf16_t* ap = &alds[wv][m * DP + t * 32 + half * 8];
    a.u[0] = *(const uint4*)ap;
    a.u[1] = *(const uint4*)(ap + 16);
    const bf16_t* bp = wzT + (t * 32 + half * 8) * NH + m;
    for (int s = 0; s < 8; s++) { b.v[s] = bp[s * NH]; b.v[8 + s] = bp[(16 + s) * NH]; }
    acc = __builtin_amdgcn_wmma_f32_16x16x32_bf16(false, a.v, false, b.v,
                                                  (short)0, acc, false, false);
  }
  // C tile: rows = pair (k offset), cols = head. 8 consecutive k per lane.
  int h = m;
  size_t base = (size_t)h * N_TOK * N_TOK + (size_t)q * N_TOK + k0 + 8 * half;
  float4 o0 = {acc[0], acc[1], acc[2], acc[3]};
  float4 o1 = {acc[4], acc[5], acc[6], acc[7]};
  *(float4*)(bias + base)     = o0;
  *(float4*)(bias + base + 4) = o1;
}

// Flash attention per (head, 16-query tile), online softmax, all matmuls WMMA.
__global__ void k_attn(const bf16_t* __restrict__ qpad, const bf16_t* __restrict__ ktp,
                       const bf16_t* __restrict__ vb, const float* __restrict__ bias,
                       const int* __restrict__ mask, float* __restrict__ obuf) {
  __shared__ __align__(16) bf16_t plds[8][16 * 32];
  int wv = threadIdx.x >> 5, lane = threadIdx.x & 31;
  int wave = blockIdx.x * 8 + wv;
  int h  = wave / (N_TOK / 16);
  int qt = wave % (N_TOK / 16);
  int q0 = qt * 16;
  int m = lane & 15, half = lane >> 4;

  Frag16 aq[2];
#pragma unroll
  for (int t = 0; t < 2; t++) {
    const bf16_t* ap = qpad + (size_t)(q0 + m) * (NH * DHP) + h * DHP + t * 32 + half * 8;
    aq[t].u[0] = *(const uint4*)ap;
    aq[t].u[1] = *(const uint4*)(ap + 16);
  }

  v8f oacc[3] = {};
  float mrow[8], lrow[8];
#pragma unroll
  for (int r = 0; r < 8; r++) { mrow[r] = -1e30f; lrow[r] = 0.0f; }

  for (int kb = 0; kb < N_TOK / 32; kb++) {
    int k0 = kb * 32;
    v8f s0 = {}, s1 = {};
#pragma unroll
    for (int t = 0; t < 2; t++) {
      const bf16_t* bp = ktp + ((size_t)h * DHP + t * 32 + half * 8) * N_TOK + k0 + m;
      Frag16 b0, b1;
      for (int s = 0; s < 8; s++) {
        b0.v[s] = bp[(size_t)s * N_TOK];      b0.v[8 + s] = bp[(size_t)(16 + s) * N_TOK];
        b1.v[s] = bp[(size_t)s * N_TOK + 16]; b1.v[8 + s] = bp[(size_t)(16 + s) * N_TOK + 16];
      }
      s0 = __builtin_amdgcn_wmma_f32_16x16x32_bf16(false, aq[t].v, false, b0.v,
                                                   (short)0, s0, false, false);
      s1 = __builtin_amdgcn_wmma_f32_16x16x32_bf16(false, aq[t].v, false, b1.v,
                                                   (short)0, s1, false, false);
    }
    // bias + mask (q pre-scaled by 1/sqrt(dh))
    int key0 = k0 + m, key1 = k0 + 16 + m;
    float mb0 = (mask[key0] > 0) ? 0.0f : -1e9f;
    float mb1 = (mask[key1] > 0) ? 0.0f : -1e9f;
#pragma unroll
    for (int r = 0; r < 8; r++) {
      size_t rowb = (size_t)h * N_TOK * N_TOK + (size_t)(q0 + r + 8 * half) * N_TOK;
      s0[r] += bias[rowb + key0] + mb0;
      s1[r] += bias[rowb + key1] + mb1;
    }
    // online softmax over this 32-key slab
#pragma unroll
    for (int r = 0; r < 8; r++) {
      float mx = rmax16(fmaxf(s0[r], s1[r]));
      float mn = fmaxf(mrow[r], mx);
      float corr = __expf(mrow[r] - mn);
      float p0 = __expf(s0[r] - mn);
      float p1 = __expf(s1[r] - mn);
      lrow[r] = lrow[r] * corr + rsum16(p0 + p1);
      mrow[r] = mn;
#pragma unroll
      for (int c = 0; c < 3; c++) oacc[c][r] *= corr;
      int qrow = r + 8 * half;
      plds[wv][qrow * 32 + m]      = (bf16_t)p0;   // C layout -> [query][key] in LDS
      plds[wv][qrow * 32 + 16 + m] = (bf16_t)p1;
    }
    __syncthreads();  // uniform; covers DS->WMMA hazard conservatively

    // P (16x32) @ V (32x48): A frag from LDS, B frags contiguous from v
    Frag16 ap;
    const bf16_t* pp = &plds[wv][m * 32 + half * 8];
    ap.u[0] = *(const uint4*)(pp);
    ap.u[1] = *(const uint4*)(pp + 16);
    const bf16_t* vp = vb + (size_t)(k0 + half * 8) * DIM + h * DH + m;
#pragma unroll
    for (int c = 0; c < 3; c++) {
      Frag16 b;
      for (int s = 0; s < 8; s++) { b.v[s] = vp[(size_t)s * DIM + c * 16]; b.v[8 + s] = vp[(size_t)(16 + s) * DIM + c * 16]; }
      oacc[c] = __builtin_amdgcn_wmma_f32_16x16x32_bf16(false, ap.v, false, b.v,
                                                        (short)0, oacc[c], false, false);
    }
    __syncthreads();
  }
#pragma unroll
  for (int r = 0; r < 8; r++) {
    float inv = 1.0f / lrow[r];
    int n = q0 + r + 8 * half;
#pragma unroll
    for (int c = 0; c < 3; c++)
      obuf[(size_t)n * DIM + h * DH + c * 16 + m] = oacc[c][r] * inv;
  }
}

__global__ void k_gate(const float* __restrict__ o, const float* __restrict__ g,
                       bf16_t* __restrict__ og) {
  int id = blockIdx.x * blockDim.x + threadIdx.x;
  og[id] = (bf16_t)(o[id] * g[id]);
}

extern "C" void kernel_launch(void* const* d_in, const int* in_sizes, int n_in,
                              void* d_out, int out_size, void* d_ws, size_t ws_size,
                              hipStream_t stream) {
  const float* s    = (const float*)d_in[0];
  const float* z    = (const float*)d_in[1];
  const int*   mask = (const int*)d_in[2];
  const float* lnsw = (const float*)d_in[3];
  const float* lnsb = (const float*)d_in[4];
  const float* wq   = (const float*)d_in[5];
  const float* bq   = (const float*)d_in[6];
  const float* wk   = (const float*)d_in[7];
  const float* wv   = (const float*)d_in[8];
  const float* lnzw = (const float*)d_in[9];
  const float* lnzb = (const float*)d_in[10];
  const float* wz   = (const float*)d_in[11];
  const float* wg   = (const float*)d_in[12];
  const float* wo   = (const float*)d_in[13];
  float* out = (float*)d_out;

  char* p = (char*)d_ws;
  auto alloc = [&](size_t bytes) -> char* {
    char* r = p; p += (bytes + 255) & ~(size_t)255; return r;
  };
  bf16_t* sn   = (bf16_t*)alloc((size_t)N_TOK * DIM * 2);
  bf16_t* wqT  = (bf16_t*)alloc((size_t)DIM * DIM * 2);
  bf16_t* wkT  = (bf16_t*)alloc((size_t)DIM * DIM * 2);
  bf16_t* wvT  = (bf16_t*)alloc((size_t)DIM * DIM * 2);
  bf16_t* wgT  = (bf16_t*)alloc((size_t)DIM * DIM * 2);
  bf16_t* woT  = (bf16_t*)alloc((size_t)DIM * DIM * 2);
  bf16_t* wzT  = (bf16_t*)alloc((size_t)DP * NH * 2);
  bf16_t* qpad = (bf16_t*)alloc((size_t)N_TOK * NH * DHP * 2);
  bf16_t* ktp  = (bf16_t*)alloc((size_t)NH * DHP * N_TOK * 2);
  bf16_t* vbuf = (bf16_t*)alloc((size_t)N_TOK * DIM * 2);
  float*  gbuf = (float*) alloc((size_t)N_TOK * DIM * 4);
  float*  obuf = (float*) alloc((size_t)N_TOK * DIM * 4);
  bf16_t* ogb  = (bf16_t*)alloc((size_t)N_TOK * DIM * 2);
  float*  bias = (float*) alloc((size_t)NH * N_TOK * N_TOK * 4);

  int tc = N_TOK * DIM;
  k_transpose_bf16<<<(tc + 255) / 256, 256, 0, stream>>>(wq, wqT, DIM, DIM);
  k_transpose_bf16<<<(tc + 255) / 256, 256, 0, stream>>>(wk, wkT, DIM, DIM);
  k_transpose_bf16<<<(tc + 255) / 256, 256, 0, stream>>>(wv, wvT, DIM, DIM);
  k_transpose_bf16<<<(tc + 255) / 256, 256, 0, stream>>>(wg, wgT, DIM, DIM);
  k_transpose_bf16<<<(tc + 255) / 256, 256, 0, stream>>>(wo, woT, DIM, DIM);
  k_transpose_bf16<<<(NH * DP + 255) / 256, 256, 0, stream>>>(wz, wzT, NH, DP);

  k_ln_s<<<N_TOK, 256, 0, stream>>>(s, lnsw, lnsb, sn);

  hipMemsetAsync(qpad, 0, (size_t)N_TOK * NH * DHP * 2, stream);
  hipMemsetAsync(ktp,  0, (size_t)NH * DHP * N_TOK * 2, stream);

  int gemmBlocks = (N_TOK / 16) * (DIM / 16) / 8;  // 288 blocks, 8 waves each
  k_gemm<<<gemmBlocks, 256, 0, stream>>>(sn, wqT, bq, qpad, nullptr, MODE_Q);
  k_gemm<<<gemmBlocks, 256, 0, stream>>>(sn, wkT, nullptr, ktp, nullptr, MODE_KT);
  k_gemm<<<gemmBlocks, 256, 0, stream>>>(sn, wvT, nullptr, vbuf, nullptr, MODE_V);
  k_gemm<<<gemmBlocks, 256, 0, stream>>>(sn, wgT, nullptr, nullptr, gbuf, MODE_G);

  k_pairbias<<<(N_TOK * (N_TOK / 16)) / 8, 256, 0, stream>>>(z, lnzw, lnzb, wzT, bias);

  k_attn<<<(NH * (N_TOK / 16)) / 8, 256, 0, stream>>>(qpad, ktp, vbuf, bias, mask, obuf);

  k_gate<<<tc / 256, 256, 0, stream>>>(obuf, gbuf, ogb);

  k_gemm<<<gemmBlocks, 256, 0, stream>>>(ogb, woT, nullptr, nullptr, out, MODE_OUT);
}
